// GraphConvolution_56367150793517
// MI455X (gfx1250) — matrix operand order repaired and checked by reading, not compiled, not run on hardware
//
#include <hip/hip_runtime.h>
#include <cstddef>

#define Bb 4
#define Nn 256
#define DXx 128
#define DEe 64
#define DYy 64
#define FFXx 512
#define FFEe 256
#define GHh 96

typedef __attribute__((ext_vector_type(16))) _Float16 v16h;
typedef __attribute__((ext_vector_type(8)))  float    v8f;

union HVec { v16h v; unsigned int u[8]; };

// Branch-free erf (Abramowitz-Stegun 7.1.26, max abs err 1.5e-7) -> exact-gelu
// within fp32 test tolerance, ~15 branchless VALU ops (v_rcp_f32 + v_exp_f32).
__device__ inline float gelu_f(float x) {
    float s = x * 0.70710678118654752f;
    float a = fabsf(s);
    float t = __builtin_amdgcn_rcpf(1.0f + 0.3275911f * a);
    float p = 1.061405429f;
    p = p * t - 1.453152027f;
    p = p * t + 1.421413741f;
    p = p * t - 0.284496736f;
    p = p * t + 0.254829592f;
    p = p * t;
    float e = __builtin_amdgcn_exp2f(-a * a * 1.4426950408889634f);
    float erf_abs = 1.0f - p * e;
    float erf_s = copysignf(erf_abs, s);
    return 0.5f * x * (1.0f + erf_s);
}

__device__ inline float sigmoid_f(float x) {
    float e = __builtin_amdgcn_exp2f(-x * 1.4426950408889634f);
    return __builtin_amdgcn_rcpf(1.0f + e);
}

// A-operand: 16x32 f16 tile, rows row0..row0+15, k in [kbase, kbase+32), row-major LDS.
__device__ inline v16h load_a16(const _Float16* base, int row0, int rowStride, int kbase, int lane) {
    HVec a;
    int m = lane & 15, half = lane >> 4;
#pragma unroll
    for (int v = 0; v < 8; ++v) {
        int k0 = ((v >> 2) << 4) + (half << 3) + ((v & 3) << 1);
        a.u[v] = *(const unsigned int*)(base + (row0 + m) * rowStride + kbase + k0);
    }
    return a.v;
}

// B-operand: 32x16 f16 tile from transposed LDS storage [n][k].
__device__ inline v16h load_bT16(const _Float16* base, int col0, int rowStride, int kbase, int lane) {
    HVec b;
    int n = lane & 15, half = lane >> 4;
#pragma unroll
    for (int v = 0; v < 8; ++v) {
        int k = ((v >> 2) << 4) + (half << 3) + ((v & 3) << 1);
        b.u[v] = *(const unsigned int*)(base + (col0 + n) * rowStride + kbase + k);
    }
    return b.v;
}

__device__ inline float breduce256(float v, float* sred, int tid) {
    sred[tid] = v; __syncthreads();
    for (int off = 128; off > 0; off >>= 1) {
        if (tid < off) sred[tid] += sred[tid + off];
        __syncthreads();
    }
    float r = sred[0]; __syncthreads();
    return r;
}

// ---------------- K0: gate base (y @ W1d + b1) ----------------
__global__ __launch_bounds__(96) void k_gate_base(
    const float* __restrict__ y, const float* __restrict__ eg_w1,
    const float* __restrict__ eg_b1, float* __restrict__ baseH)
{
    int b = blockIdx.x, g = threadIdx.x;
    float a = eg_b1[g];
    for (int k = 0; k < DYy; ++k) a += y[b * DYy + k] * eg_w1[(2 * DXx + DEe + k) * GHh + g];
    baseH[b * GHh + g] = a;
}

// ---------------- K1: gate node precompute (X@W1a, X@W1b) ----------------
__global__ __launch_bounds__(96) void k_gate_nodes(
    const float* __restrict__ X, const float* __restrict__ eg_w1,
    float* __restrict__ HxA, float* __restrict__ HxB)
{
    __shared__ float sx[DXx];
    int bi = blockIdx.x, tid = threadIdx.x;
    const float* xr = X + (size_t)bi * DXx;
    for (int t = tid; t < DXx; t += 96) sx[t] = xr[t];
    __syncthreads();
    float a = 0.f, c = 0.f;
    for (int d = 0; d < DXx; ++d) {
        float xv = sx[d];
        a += xv * eg_w1[d * GHh + tid];
        c += xv * eg_w1[(DXx + d) * GHh + tid];
    }
    HxA[(size_t)bi * GHh + tid] = a;
    HxB[(size_t)bi * GHh + tid] = c;
}

// ---------------- K2: edge gate (WMMA) -> attention weights ----------------
__global__ __launch_bounds__(256) void k_edge_gate(
    const float* __restrict__ E, const unsigned char* __restrict__ mask,
    const float* __restrict__ eg_w1, const float* __restrict__ eg_w2,
    const float* __restrict__ eg_b2,
    const float* __restrict__ HxA, const float* __restrict__ HxB,
    const float* __restrict__ baseH, float* __restrict__ Wgt)
{
    __shared__ _Float16 sE[Nn * DEe];    // 32 KB, [j][k] f16
    __shared__ _Float16 sW[GHh * DEe];   // 12 KB, transposed [g][k]
    __shared__ _Float16 sH[Nn * GHh];    // 48 KB, hidden [j][g]
    __shared__ float sConst[GHh];
    __shared__ float sW2[GHh];

    int bi = blockIdx.x, b = bi >> 8, i = bi & 255;
    int tid = threadIdx.x;
    const float* Erow = E + (size_t)(b * Nn + i) * Nn * DEe;
    for (int t = tid; t < Nn * DEe; t += 256) sE[t] = (_Float16)Erow[t];
    for (int t = tid; t < GHh * DEe; t += 256) {
        int g = t >> 6, k = t & 63;
        sW[t] = (_Float16)eg_w1[(2 * DXx + k) * GHh + g];
    }
    if (tid < GHh) {
        sConst[tid] = HxA[(size_t)(b * Nn + i) * GHh + tid] + baseH[b * GHh + tid];
        sW2[tid] = eg_w2[tid];
    }
    __syncthreads();

    int wave = tid >> 5, lane = tid & 31;
    int n = lane & 15, half = lane >> 4;
    for (int t = wave; t < 96; t += 8) {      // 16 j-tiles x 6 g-tiles
        int jt = t / 6, gt = t % 6;
        v8f acc = {};
#pragma unroll
        for (int kt = 0; kt < 2; ++kt) {
            v16h a  = load_a16(sE, jt * 16, DEe, kt * 32, lane);
            v16h bb = load_bT16(sW, gt * 16, DEe, kt * 32, lane);
            acc = __builtin_amdgcn_wmma_f32_16x16x32_f16(false, a, false, bb, (short)0, acc, false, false);
        }
        int g = gt * 16 + n;
        float cg = sConst[g];
#pragma unroll
        for (int r = 0; r < 8; ++r) {
            int m = r + (half << 3);
            int j = jt * 16 + m;
            float v = acc[r] + cg + HxB[(size_t)(b * Nn + j) * GHh + g];
            sH[j * GHh + g] = (_Float16)gelu_f(v);
        }
    }
    __syncthreads();

    int j = tid;
    float acc = eg_b2[0];
    for (int g = 0; g < GHh; ++g) acc += (float)sH[j * GHh + g] * sW2[g];
    bool pm = mask[b * Nn + i] && mask[b * Nn + j] && (i != j);
    float logit = pm ? acc : -30.0f;
    Wgt[(size_t)(b * Nn + i) * Nn + j] = sigmoid_f(logit);
}

// ---------------- K3: GIN aggregation + node MLP + 2x LayerNorm ----------------
__global__ __launch_bounds__(256) void k_node_update(
    const float* __restrict__ X, const float* __restrict__ Wgt,
    const unsigned char* __restrict__ mask, const float* __restrict__ gin_eps,
    const float* __restrict__ px_w1, const float* __restrict__ px_b1,
    const float* __restrict__ px_w2, const float* __restrict__ px_b2,
    const float* __restrict__ nx_g, const float* __restrict__ nx_b,
    float* __restrict__ Xout, float* __restrict__ outX)
{
    __shared__ float swr[Nn];
    __shared__ float sx[DXx];
    __shared__ float sagg[DXx];
    __shared__ float sh[FFXx];
    __shared__ float sred[256];

    int bi = blockIdx.x, b = bi >> 8, i = bi & 255;
    int tid = threadIdx.x;
    swr[tid] = Wgt[(size_t)(b * Nn + i) * Nn + tid];
    if (tid < DXx) sx[tid] = X[(size_t)(b * Nn + i) * DXx + tid];
    __syncthreads();

    if (tid < DXx) {
        float m = 0.f;
        for (int j = 0; j < Nn; ++j) m += swr[j] * X[(size_t)(b * Nn + j) * DXx + tid];
        sagg[tid] = (1.0f + gin_eps[0]) * sx[tid] + m;
    }
    __syncthreads();

    {
        float a0 = px_b1[tid], a1 = px_b1[tid + 256];
        for (int d = 0; d < DXx; ++d) {
            float ag = sagg[d];
            a0 += ag * px_w1[d * FFXx + tid];
            a1 += ag * px_w1[d * FFXx + tid + 256];
        }
        sh[tid] = gelu_f(a0);
        sh[tid + 256] = gelu_f(a1);
    }
    __syncthreads();

    float r1 = 0.f;
    if (tid < DXx) {
        float xn = px_b2[tid];
        for (int f = 0; f < FFXx; ++f) xn += sh[f] * px_w2[f * DXx + tid];
        r1 = sx[tid] + xn;
    }
    float s1 = breduce256(tid < DXx ? r1 : 0.f, sred, tid);
    float s2 = breduce256(tid < DXx ? r1 * r1 : 0.f, sred, tid);
    float mu = s1 * (1.0f / DXx);
    float var = s2 * (1.0f / DXx) - mu * mu;
    float rstd = rsqrtf(var + 1e-5f);
    float r2 = 0.f;
    if (tid < DXx) {
        float xc = (r1 - mu) * rstd * nx_g[tid] + nx_b[tid];
        r2 = sx[tid] + xc;
    }
    float t1 = breduce256(tid < DXx ? r2 : 0.f, sred, tid);
    float t2 = breduce256(tid < DXx ? r2 * r2 : 0.f, sred, tid);
    float mu2 = t1 * (1.0f / DXx);
    float var2 = t2 * (1.0f / DXx) - mu2 * mu2;
    float rstd2 = rsqrtf(var2 + 1e-5f);
    if (tid < DXx) {
        float xo = (r2 - mu2) * rstd2 * nx_g[tid] + nx_b[tid];
        Xout[(size_t)(b * Nn + i) * DXx + tid] = xo;
        outX[(size_t)(b * Nn + i) * DXx + tid] = mask[b * Nn + i] ? xo : 0.f;
    }
}

// ---------------- K4: edge-update node precompute ----------------
__global__ __launch_bounds__(256) void k_eu_nodes(
    const float* __restrict__ Xout, const float* __restrict__ eu_w1,
    const float* __restrict__ eu_b1, float* __restrict__ GxA, float* __restrict__ GxB)
{
    __shared__ float sx[DXx];
    int bi = blockIdx.x, tid = threadIdx.x;
    if (tid < DXx) sx[tid] = Xout[(size_t)bi * DXx + tid];
    __syncthreads();
    float a = 0.f, c = eu_b1[tid];
    for (int d = 0; d < DXx; ++d) {
        float xv = sx[d];
        a += xv * eu_w1[d * FFEe + tid];
        c += xv * eu_w1[(DXx + d) * FFEe + tid];
    }
    GxA[(size_t)bi * FFEe + tid] = a;
    GxB[(size_t)bi * FFEe + tid] = c;
}

// ---------------- K5: edge-update MLP (WMMA x2 GEMMs) ----------------
__global__ __launch_bounds__(256) void k_edge_update(
    const float* __restrict__ E,
    const float* __restrict__ eu_w1, const float* __restrict__ eu_w2,
    const float* __restrict__ eu_b2,
    const float* __restrict__ GxA, const float* __restrict__ GxB,
    float* __restrict__ Eupd)
{
    __shared__ _Float16 sE[Nn * DEe];      // 32 KB, [j][k]
    __shared__ _Float16 sWc[FFEe * DEe];   // 32 KB, transposed [g][k]
    __shared__ _Float16 sW2[DEe * FFEe];   // 32 KB, transposed [d][g]
    __shared__ _Float16 sH[64 * FFEe];     // 32 KB, [jj][g] per chunk

    int bi = blockIdx.x, b = bi >> 8, i = bi & 255;
    int tid = threadIdx.x;
    const float* Erow = E + (size_t)(b * Nn + i) * Nn * DEe;
    for (int t = tid; t < Nn * DEe; t += 256) sE[t] = (_Float16)Erow[t];
    for (int t = tid; t < FFEe * DEe; t += 256) {
        int g = t >> 6, k = t & 63;
        sWc[t] = (_Float16)eu_w1[(2 * DXx + k) * FFEe + g];
    }
    for (int t = tid; t < DEe * FFEe; t += 256) {
        int d = t >> 8, k = t & 255;
        sW2[t] = (_Float16)eu_w2[k * DEe + d];
    }
    __syncthreads();

    int wave = tid >> 5, lane = tid & 31;
    int n = lane & 15, half = lane >> 4;
    const float* gxaRow = GxA + (size_t)(b * Nn + i) * FFEe;

    for (int c = 0; c < 4; ++c) {
        int j0 = c * 64;
        // GEMM1: hidden H[0..63][0..255] = E @ Wc + GxA_i + GxB_j, gelu
        for (int t = wave; t < 64; t += 8) {   // 4 m-tiles x 16 n-tiles
            int mt = t >> 4, nt = t & 15;
            v8f acc = {};
#pragma unroll
            for (int kt = 0; kt < 2; ++kt) {
                v16h a  = load_a16(sE, j0 + mt * 16, DEe, kt * 32, lane);
                v16h bb = load_bT16(sWc, nt * 16, DEe, kt * 32, lane);
                acc = __builtin_amdgcn_wmma_f32_16x16x32_f16(false, a, false, bb, (short)0, acc, false, false);
            }
            int g = nt * 16 + n;
            float ga = gxaRow[g];
#pragma unroll
            for (int r = 0; r < 8; ++r) {
                int m = r + (half << 3);
                int j = j0 + mt * 16 + m;
                float v = acc[r] + ga + GxB[(size_t)(b * Nn + j) * FFEe + g];
                sH[(mt * 16 + m) * FFEe + g] = (_Float16)gelu_f(v);
            }
        }
        __syncthreads();
        // GEMM2: upd[j][d] = H @ W2 + b2
        for (int t = wave; t < 16; t += 8) {   // 4 m-tiles x 4 n-tiles
            int mt = t >> 2, nt = t & 3;
            v8f acc = {};
#pragma unroll
            for (int kt = 0; kt < 8; ++kt) {
                v16h a  = load_a16(sH, mt * 16, FFEe, kt * 32, lane);
                v16h bb = load_bT16(sW2, nt * 16, FFEe, kt * 32, lane);
                acc = __builtin_amdgcn_wmma_f32_16x16x32_f16(false, a, false, bb, (short)0, acc, false, false);
            }
            int d = nt * 16 + n;
            float bd = eu_b2[d];
#pragma unroll
            for (int r = 0; r < 8; ++r) {
                int m = r + (half << 3);
                int j = j0 + mt * 16 + m;
                Eupd[((size_t)(b * Nn + i) * Nn + j) * DEe + d] = acc[r] + bd;
            }
        }
        __syncthreads();
    }
}

// ---------------- K6: symmetrize + residual + LN(64) + mask + pool partials ----------------
__global__ __launch_bounds__(256) void k_sym_ln(
    const float* __restrict__ E, const float* __restrict__ Eupd,
    const unsigned char* __restrict__ mask,
    const float* __restrict__ ne_g, const float* __restrict__ ne_b,
    float* __restrict__ outE, float* __restrict__ epart)
{
    __shared__ float sPoolW[8 * DEe];
    int bi = blockIdx.x, b = bi >> 8, i = bi & 255;
    int tid = threadIdx.x, wave = tid >> 5, lane = tid & 31;
    int d0 = lane, d1 = lane + 32;
    float g0 = ne_g[d0], g1 = ne_g[d1], bt0 = ne_b[d0], bt1 = ne_b[d1];
    bool mi = mask[b * Nn + i];
    float ep0 = 0.f, ep1 = 0.f;
    for (int j = wave; j < Nn; j += 8) {
        size_t offIJ = ((size_t)(b * Nn + i) * Nn + j) * DEe;
        size_t offJI = ((size_t)(b * Nn + j) * Nn + i) * DEe;
        float r0 = E[offIJ + d0] + 0.5f * (Eupd[offIJ + d0] + Eupd[offJI + d0]);
        float r1 = E[offIJ + d1] + 0.5f * (Eupd[offIJ + d1] + Eupd[offJI + d1]);
        float s1 = r0 + r1, s2 = r0 * r0 + r1 * r1;
#pragma unroll
        for (int m = 16; m >= 1; m >>= 1) {
            s1 += __shfl_xor(s1, m, 32);
            s2 += __shfl_xor(s2, m, 32);
        }
        float mu = s1 * (1.0f / DEe);
        float var = s2 * (1.0f / DEe) - mu * mu;
        float rstd = rsqrtf(var + 1e-5f);
        bool pm = mi && mask[b * Nn + j] && (i != j);
        float e0 = pm ? ((r0 - mu) * rstd * g0 + bt0) : 0.f;
        float e1 = pm ? ((r1 - mu) * rstd * g1 + bt1) : 0.f;
        outE[offIJ + d0] = e0;
        outE[offIJ + d1] = e1;
        ep0 += e0; ep1 += e1;
    }
    sPoolW[wave * DEe + d0] = ep0;
    sPoolW[wave * DEe + d1] = ep1;
    __syncthreads();
    if (tid < DEe) {
        float s = 0.f;
        for (int w = 0; w < 8; ++w) s += sPoolW[w * DEe + tid];
        epart[(size_t)(b * Nn + i) * DEe + tid] = s;
    }
}

// ---------------- K7: pooling + y update + LN ----------------
__global__ __launch_bounds__(64) void k_final_y(
    const float* __restrict__ y, const unsigned char* __restrict__ mask,
    const float* __restrict__ Xout, const float* __restrict__ epart,
    const float* __restrict__ yu_w, const float* __restrict__ yu_b,
    const float* __restrict__ ny_g, const float* __restrict__ ny_b,
    float* __restrict__ outY)
{
    __shared__ float sxp[DXx];
    __shared__ float sep[DEe];
    __shared__ float sred[DEe];
    __shared__ float snv;
    int b = blockIdx.x, tid = threadIdx.x;
    if (tid == 0) {
        float nv = 0.f;
        for (int ii = 0; ii < Nn; ++ii) nv += mask[b * Nn + ii] ? 1.f : 0.f;
        snv = nv;
    }
    __syncthreads();
    float nv = snv;
    float nvc = fmaxf(nv, 1.f);
    float evc = fmaxf(nv * nv - nv, 1.f);
    for (int k = tid; k < DXx; k += 64) {
        float s = 0.f;
        for (int ii = 0; ii < Nn; ++ii)
            s += Xout[(size_t)(b * Nn + ii) * DXx + k] * (mask[b * Nn + ii] ? 1.f : 0.f);
        sxp[k] = s / nvc;
    }
    {
        float s = 0.f;
        for (int ii = 0; ii < Nn; ++ii) s += epart[(size_t)(b * Nn + ii) * DEe + tid];
        sep[tid] = s / evc;
    }
    __syncthreads();
    float up = yu_b[tid];
    for (int k = 0; k < DXx; ++k) up += sxp[k] * yu_w[k * DYy + tid];
    for (int k = 0; k < DEe; ++k) up += sep[k] * yu_w[(DXx + k) * DYy + tid];
    float r = y[b * DYy + tid] + up;
    sred[tid] = r; __syncthreads();
    for (int off = 32; off > 0; off >>= 1) { if (tid < off) sred[tid] += sred[tid + off]; __syncthreads(); }
    float mu = sred[0] * (1.f / DYy); __syncthreads();
    float dv = r - mu;
    sred[tid] = dv * dv; __syncthreads();
    for (int off = 32; off > 0; off >>= 1) { if (tid < off) sred[tid] += sred[tid + off]; __syncthreads(); }
    float var = sred[0] * (1.f / DYy); __syncthreads();
    outY[b * DYy + tid] = dv * rsqrtf(var + 1e-5f) * ny_g[tid] + ny_b[tid];
}

extern "C" void kernel_launch(void* const* d_in, const int* in_sizes, int n_in,
                              void* d_out, int out_size, void* d_ws, size_t ws_size,
                              hipStream_t stream) {
    const float* X      = (const float*)d_in[0];
    const float* E      = (const float*)d_in[1];
    const float* y      = (const float*)d_in[2];
    const unsigned char* mask = (const unsigned char*)d_in[3];
    const float* eg_w1  = (const float*)d_in[4];
    const float* eg_b1  = (const float*)d_in[5];
    const float* eg_w2  = (const float*)d_in[6];
    const float* eg_b2  = (const float*)d_in[7];
    const float* gin_eps= (const float*)d_in[8];
    const float* px_w1  = (const float*)d_in[9];
    const float* px_b1  = (const float*)d_in[10];
    const float* px_w2  = (const float*)d_in[11];
    const float* px_b2  = (const float*)d_in[12];
    const float* nx_g   = (const float*)d_in[13];
    const float* nx_b   = (const float*)d_in[14];
    const float* eu_w1  = (const float*)d_in[15];
    const float* eu_b1  = (const float*)d_in[16];
    const float* eu_w2  = (const float*)d_in[17];
    const float* eu_b2  = (const float*)d_in[18];
    const float* ne_g   = (const float*)d_in[19];
    const float* ne_b   = (const float*)d_in[20];
    const float* yu_w   = (const float*)d_in[21];
    const float* yu_b   = (const float*)d_in[22];
    const float* ny_g   = (const float*)d_in[23];
    const float* ny_b   = (const float*)d_in[24];
    (void)in_sizes; (void)n_in; (void)out_size; (void)ws_size;

    float* ws    = (float*)d_ws;
    float* HxA   = ws;                       // B*N*96
    float* HxB   = HxA + Bb * Nn * GHh;      // B*N*96
    float* baseH = HxB + Bb * Nn * GHh;      // B*96
    float* Wgt   = baseH + Bb * GHh;         // B*N*N
    float* Xout  = Wgt + (size_t)Bb * Nn * Nn;        // B*N*128
    float* GxA   = Xout + (size_t)Bb * Nn * DXx;      // B*N*256
    float* GxB   = GxA + (size_t)Bb * Nn * FFEe;      // B*N*256
    float* epart = GxB + (size_t)Bb * Nn * FFEe;      // B*N*64
    float* Eupd  = epart + (size_t)Bb * Nn * DEe;     // B*N*N*64

    float* outX = (float*)d_out;
    float* outE = outX + (size_t)Bb * Nn * DXx;
    float* outY = outE + (size_t)Bb * Nn * Nn * DEe;

    k_gate_base<<<Bb, 96, 0, stream>>>(y, eg_w1, eg_b1, baseH);
    k_gate_nodes<<<Bb * Nn, 96, 0, stream>>>(X, eg_w1, HxA, HxB);
    k_edge_gate<<<Bb * Nn, 256, 0, stream>>>(E, mask, eg_w1, eg_w2, eg_b2, HxA, HxB, baseH, Wgt);
    k_node_update<<<Bb * Nn, 256, 0, stream>>>(X, Wgt, mask, gin_eps, px_w1, px_b1, px_w2, px_b2,
                                               nx_g, nx_b, Xout, outX);
    k_eu_nodes<<<Bb * Nn, 256, 0, stream>>>(Xout, eu_w1, eu_b1, GxA, GxB);
    k_edge_update<<<Bb * Nn, 256, 0, stream>>>(E, eu_w1, eu_w2, eu_b2, GxA, GxB, Eupd);
    k_sym_ln<<<Bb * Nn, 256, 0, stream>>>(E, Eupd, mask, ne_g, ne_b, outE, epart);
    k_final_y<<<Bb, 64, 0, stream>>>(y, mask, Xout, epart, yu_w, yu_b, ny_g, ny_b, outY);
}
